// VariedSizePatchAttention_40106404610310
// MI455X (gfx1250) — compile-verified, matrix operand order checked by reading, not run on hardware
//
#include <hip/hip_runtime.h>
#include <hip/hip_bf16.h>
#include <math.h>

typedef __attribute__((ext_vector_type(16))) _Float16 v16h;
typedef __attribute__((ext_vector_type(8)))  float    v8f;

#define IMG   384
#define PP    4
#define CIN   6
#define D0    96
#define BATCH 2
#define LN_EPS 1e-5f
#define BN_EPS 1e-5f

__device__ __forceinline__ float gelu_exact(float x) {
    return 0.5f * x * (1.0f + erff(x * 0.70710678118654752f));
}
__device__ __forceinline__ int clampi(int v, int lo, int hi) {
    return v < lo ? lo : (v > hi ? hi : v);
}

// ---------------------------------------------------------------------------
// Patch embed: 1x1 conv CIN->D0. Emits NCHW f32 (for bilinear sampling) and
// NHWC f16 (vectorizable im2col operand for the WMMA conv) in one pass.
// ---------------------------------------------------------------------------
__global__ void k_embed(const float* __restrict__ x, const float* __restrict__ w,
                        const float* __restrict__ b, float* __restrict__ xe,
                        _Float16* __restrict__ xen) {
    const int HW = IMG * IMG;
    int i = blockIdx.x * blockDim.x + threadIdx.x;
    if (i >= BATCH * HW) return;
    int bb = i / HW, hw = i % HW;
    float in[CIN];
    #pragma unroll
    for (int c = 0; c < CIN; ++c) in[c] = x[((size_t)bb * CIN + c) * HW + hw];
    for (int d = 0; d < D0; ++d) {
        float acc = b[d];
        #pragma unroll
        for (int c = 0; c < CIN; ++c) acc += in[c] * w[d * CIN + c];
        xe[((size_t)bb * D0 + d) * HW + hw] = acc;
        xen[((size_t)bb * HW + hw) * D0 + d] = (_Float16)acc;
    }
}

// Permute ph1 weights (d, c, ky, kx) -> (d, tap, c), f32 -> f16.
#define CK (D0 * 9)   // 864
__global__ void k_permw(const float* __restrict__ w, _Float16* __restrict__ wp) {
    int i = blockIdx.x * blockDim.x + threadIdx.x;
    if (i >= D0 * CK) return;
    int d = i / CK; int k = i % CK;
    int t9 = k / D0; int c = k % D0;
    int ky = t9 / 3, kx = t9 % 3;
    wp[i] = (_Float16)w[(((size_t)d * D0 + c) * 3 + ky) * 3 + kx];
}

// ---------------------------------------------------------------------------
// ph1: 3x3 conv D0->D0 pad 1 + GELU as implicit-GEMM WMMA.
// M = B*HW pixel rows (multiple of 128), N = 96 out-channels, K = 9*96 = 864
// ordered (tap-major, channel-minor) so each 16-half run is one contiguous
// 32B NHWC load at a shifted pixel.
// ---------------------------------------------------------------------------
#define CBM 128
#define CBK 32
#define CLDA 40

__global__ __launch_bounds__(256)
void k_ph1_wmma(const _Float16* __restrict__ xen, const _Float16* __restrict__ w16,
                const float* __restrict__ bias, float* __restrict__ h1)
{
    __shared__ _Float16 As[CBM * CLDA];   // 128 x 32 (+pad)
    __shared__ _Float16 Bs[D0 * CLDA];    // 96 x 32 (+pad)

    const int tid   = threadIdx.x;
    const int lane  = tid & 31;
    const int wid   = tid >> 5;
    const int wm    = wid & 3;        // 4 waves along M (32 rows each)
    const int wn    = wid >> 2;       // 2 waves along N (48 cols each)
    const int lm    = lane & 15;
    const int lhalf = lane >> 4;
    const int klane = lhalf * 8;

    const int HW = IMG * IMG;
    const int bm = blockIdx.x * CBM;

    const int ldRow = tid >> 1;
    const int ldK   = (tid & 1) * 16;

    // pixel coordinates for this thread's im2col row
    int grow = bm + ldRow;
    int bb = grow / HW;
    int hw = grow % HW;
    int yy = hw / IMG;
    int xx = hw % IMG;
    const _Float16* xbase = xen + (size_t)bb * HW * D0;

    v8f acc[2][3] = {};

    for (int kt = 0; kt < CK; kt += CBK) {
        // --- A tile: two contiguous 16B NHWC loads per thread (halo -> 0) ---
        {
            int k0 = kt + ldK;          // 16-aligned, never crosses a tap
            int t9 = k0 / D0;
            int c0 = k0 - t9 * D0;
            int ky = t9 / 3;
            int kx = t9 - ky * 3;
            int sy = yy + ky - 1;
            int sx = xx + kx - 1;
            uint4 v0 = make_uint4(0, 0, 0, 0), v1 = make_uint4(0, 0, 0, 0);
            if (sy >= 0 && sy < IMG && sx >= 0 && sx < IMG) {
                const uint4* src = (const uint4*)(xbase + ((size_t)sy * IMG + sx) * D0 + c0);
                v0 = src[0]; v1 = src[1];
            }
            uint4* dst = (uint4*)(&As[ldRow * CLDA + ldK]);
            dst[0] = v0; dst[1] = v1;
        }
        // --- B tile: contiguous permuted-weight rows ---
        if (ldRow < D0) {
            const uint4* src = (const uint4*)(w16 + (size_t)ldRow * CK + kt + ldK);
            uint4* dst = (uint4*)(&Bs[ldRow * CLDA + ldK]);
            dst[0] = src[0];
            dst[1] = src[1];
        }
        __syncthreads();

        v16h af[2], bf[3];
        #pragma unroll
        for (int mi = 0; mi < 2; ++mi) {
            const _Float16* ap = &As[(wm * 32 + mi * 16 + lm) * CLDA];
            #pragma unroll
            for (int j = 0; j < 16; ++j) {
                int kk = klane + j + ((j >= 8) ? 8 : 0);
                af[mi][j] = ap[kk];
            }
        }
        #pragma unroll
        for (int ni = 0; ni < 3; ++ni) {
            const _Float16* bp = &Bs[(wn * 48 + ni * 16 + lm) * CLDA];
            #pragma unroll
            for (int j = 0; j < 16; ++j) {
                int kk = klane + j + ((j >= 8) ? 8 : 0);
                bf[ni][j] = bp[kk];
            }
        }

        #pragma unroll
        for (int mi = 0; mi < 2; ++mi)
            #pragma unroll
            for (int ni = 0; ni < 3; ++ni)
                acc[mi][ni] = __builtin_amdgcn_wmma_f32_16x16x32_f16(
                    false, af[mi], false, bf[ni], (short)0, acc[mi][ni], false, false);

        __syncthreads();
    }

    // --- epilogue: bias + GELU, store transposed into NCHW h1 ---
    #pragma unroll
    for (int mi = 0; mi < 2; ++mi) {
        #pragma unroll
        for (int ni = 0; ni < 3; ++ni) {
            int col = wn * 48 + ni * 16 + lm;   // out channel (< 96 always)
            float bcol = bias[col];
            #pragma unroll
            for (int r = 0; r < 8; ++r) {
                int row = bm + wm * 32 + mi * 16 + 8 * lhalf + r;
                int b2 = row / HW;
                int hw2 = row % HW;
                float v = gelu_exact(acc[mi][ni][r] + bcol);
                h1[((size_t)b2 * D0 + col) * HW + hw2] = v;
            }
        }
    }
}

// 4x4 stride-4 conv D0->3
__global__ void k_ph2(const float* __restrict__ h1, const float* __restrict__ w,
                      const float* __restrict__ b, float* __restrict__ pr) {
    const int HP = IMG / PP;
    const int HW = IMG * IMG;
    int i = blockIdx.x * blockDim.x + threadIdx.x;
    int total = BATCH * 3 * HP * HP;
    if (i >= total) return;
    int px = i % HP; int t = i / HP;
    int py = t % HP; t /= HP;
    int o = t % 3;  int bb = t / 3;
    float acc = b[o];
    for (int c = 0; c < D0; ++c) {
        const float* hp = h1 + ((size_t)bb * D0 + c) * HW + (size_t)(py * PP) * IMG + px * PP;
        const float* wp = w + ((size_t)o * D0 + c) * 16;
        #pragma unroll
        for (int ky = 0; ky < PP; ++ky)
            #pragma unroll
            for (int kx = 0; kx < PP; ++kx)
                acc += hp[ky * IMG + kx] * wp[ky * PP + kx];
    }
    pr[i] = acc;
}

// Deformable bilinear token sampling + LayerNorm. One block per (b,n), 96 threads.
__global__ __launch_bounds__(96)
void k_tokens(const float* __restrict__ xe, const float* __restrict__ pr,
              const float* __restrict__ lng, const float* __restrict__ lnb,
              float* __restrict__ tokens) {
    const int HP = IMG / PP, N = HP * HP, HW = IMG * IMG;
    int n = blockIdx.x % N; int bb = blockIdx.x / N;
    int d = threadIdx.x;
    int py = n / HP, px = n % HP;
    size_t prb = (size_t)bb * 3 * N;
    float dx = pr[prb + 0 * N + n];
    float dy = pr[prb + 1 * N + n];
    float ls = pr[prb + 2 * N + n];
    const float alpha = PP * 0.5f;
    float mx = alpha * tanhf(dx);
    float my = alpha * tanhf(dy);
    float s = expf(tanhf(ls));
    s = fminf(fmaxf(s, 0.5f), 2.0f);
    float cx = px * (float)PP + PP * 0.5f;
    float cy = py * (float)PP + PP * 0.5f;
    float half = s * (PP * 0.5f);
    float x1 = fminf(fmaxf(cx + mx - half, 0.f), IMG - 1.f);
    float y1 = fminf(fmaxf(cy + my - half, 0.f), IMG - 1.f);
    float x2 = fminf(fmaxf(cx + mx + half, 0.f), IMG - 1.f);
    float y2 = fminf(fmaxf(cy + my + half, 0.f), IMG - 1.f);
    const float* feat = xe + ((size_t)bb * D0 + d) * HW;
    float acc = 0.f;
    #pragma unroll
    for (int iy = 0; iy < PP; ++iy) {
        float ty = (float)iy / (PP - 1);
        float gy = y1 + (y2 - y1) * ty;
        float y0f = floorf(gy);
        float wy = gy - y0f;
        int y0 = clampi((int)y0f, 0, IMG - 1);
        int y1i = clampi((int)y0f + 1, 0, IMG - 1);
        #pragma unroll
        for (int ix = 0; ix < PP; ++ix) {
            float tx = (float)ix / (PP - 1);
            float gx = x1 + (x2 - x1) * tx;
            float x0f = floorf(gx);
            float wx = gx - x0f;
            int x0 = clampi((int)x0f, 0, IMG - 1);
            int x1i = clampi((int)x0f + 1, 0, IMG - 1);
            float v00 = feat[y0 * IMG + x0];
            float v01 = feat[y0 * IMG + x1i];
            float v10 = feat[y1i * IMG + x0];
            float v11 = feat[y1i * IMG + x1i];
            acc += (1.f - wx) * (1.f - wy) * v00 + wx * (1.f - wy) * v01 +
                   (1.f - wx) * wy * v10 + wx * wy * v11;
        }
    }
    acc *= (1.0f / (PP * PP));
    __shared__ float sh[D0];
    sh[d] = acc;
    __syncthreads();
    float mu = 0.f;
    for (int j = 0; j < D0; ++j) mu += sh[j];
    mu *= (1.0f / D0);
    float var = 0.f;
    for (int j = 0; j < D0; ++j) { float df = sh[j] - mu; var += df * df; }
    var *= (1.0f / D0);
    float outv = (acc - mu) * rsqrtf(var + LN_EPS) * lng[d] + lnb[d];
    tokens[((size_t)bb * N + n) * D0 + d] = outv;
}

// ---------------------------------------------------------------------------
// 3x3 max & avg pool over token grid -> f16 GEMM operand (M x 2d)
// ---------------------------------------------------------------------------
__global__ void k_pool(const float* __restrict__ tokens, _Float16* __restrict__ h2,
                       int Hh, int Ww, int d) {
    int N = Hh * Ww;
    long long i = (long long)blockIdx.x * blockDim.x + threadIdx.x;
    long long total = (long long)BATCH * N * d;
    if (i >= total) return;
    int c = (int)(i % d); long long t = i / d;
    int n = (int)(t % N); int bb = (int)(t / N);
    int yy = n / Ww, xx = n % Ww;
    float mx = -INFINITY, sm = 0.f;
    for (int ky = -1; ky <= 1; ++ky) {
        int sy = yy + ky; if (sy < 0 || sy >= Hh) continue;
        for (int kx = -1; kx <= 1; ++kx) {
            int sx = xx + kx; if (sx < 0 || sx >= Ww) continue;
            float v = tokens[((size_t)bb * N + sy * Ww + sx) * d + c];
            mx = fmaxf(mx, v); sm += v;
        }
    }
    size_t row = (size_t)bb * N + n;
    h2[row * (2 * d) + c]     = (_Float16)mx;
    h2[row * (2 * d) + d + c] = (_Float16)(sm * (1.0f / 9.0f));
}

__global__ void k_f2h(const float* __restrict__ in, _Float16* __restrict__ out, long long n) {
    long long i = (long long)blockIdx.x * blockDim.x + threadIdx.x;
    if (i < n) out[i] = (_Float16)in[i];
}

__global__ void k_bnprep(const float* __restrict__ g, const float* __restrict__ b,
                         const float* __restrict__ m, const float* __restrict__ v,
                         float* __restrict__ scale, float* __restrict__ shift, int d) {
    int i = blockIdx.x * blockDim.x + threadIdx.x;
    if (i < d) {
        float sc = g[i] * rsqrtf(v[i] + BN_EPS);
        scale[i] = sc;
        shift[i] = b[i] - m[i] * sc;
    }
}

// LayerNorm over last dim; optional residual add; optional f32 and f16 outputs.
__global__ __launch_bounds__(256)
void k_ln(const float* __restrict__ xin, const float* __restrict__ resid,
          const float* __restrict__ g, const float* __restrict__ b,
          float* __restrict__ out, _Float16* __restrict__ outh, int D) {
    int row = blockIdx.x;
    const float* xp = xin + (size_t)row * D;
    float s = 0.f, ss = 0.f;
    for (int j = threadIdx.x; j < D; j += blockDim.x) { float v = xp[j]; s += v; ss += v * v; }
    __shared__ float sh1[256], sh2[256];
    sh1[threadIdx.x] = s; sh2[threadIdx.x] = ss;
    __syncthreads();
    for (int off = 128; off > 0; off >>= 1) {
        if ((int)threadIdx.x < off) {
            sh1[threadIdx.x] += sh1[threadIdx.x + off];
            sh2[threadIdx.x] += sh2[threadIdx.x + off];
        }
        __syncthreads();
    }
    float mu = sh1[0] / D;
    float var = sh2[0] / D - mu * mu;
    float inv = rsqrtf(var + LN_EPS);
    for (int j = threadIdx.x; j < D; j += blockDim.x) {
        float v = (xp[j] - mu) * inv * g[j] + b[j];
        if (resid) v += resid[(size_t)row * D + j];
        if (out)  out[(size_t)row * D + j] = v;
        if (outh) outh[(size_t)row * D + j] = (_Float16)v;
    }
}

// Patch-merge 2x2 gather + LayerNorm -> f16 GEMM operand (M/4 x 4C)
__global__ __launch_bounds__(256)
void k_merge_ln(const float* __restrict__ tokens, const float* __restrict__ g,
                const float* __restrict__ b, _Float16* __restrict__ outh,
                int Hh, int Ww, int C) {
    int C4 = 4 * C;
    int W2 = Ww / 2;
    int N2 = (Hh / 2) * W2;
    int row = blockIdx.x;
    int bb = row / N2; int n2 = row % N2;
    int yy = n2 / W2, xx = n2 % W2;
    __shared__ float shv[1536];   // max 4*384
    for (int j = threadIdx.x; j < C4; j += blockDim.x) {
        int q = j / C, c = j % C;
        int sy = 2 * yy + ((q == 1 || q == 3) ? 1 : 0);
        int sx = 2 * xx + ((q >= 2) ? 1 : 0);
        shv[j] = tokens[(((size_t)bb * Hh + sy) * Ww + sx) * C + c];
    }
    __syncthreads();
    float s = 0.f, ss = 0.f;
    for (int j = threadIdx.x; j < C4; j += blockDim.x) { float v = shv[j]; s += v; ss += v * v; }
    __shared__ float sh1[256], sh2[256];
    sh1[threadIdx.x] = s; sh2[threadIdx.x] = ss;
    __syncthreads();
    for (int off = 128; off > 0; off >>= 1) {
        if ((int)threadIdx.x < off) {
            sh1[threadIdx.x] += sh1[threadIdx.x + off];
            sh2[threadIdx.x] += sh2[threadIdx.x + off];
        }
        __syncthreads();
    }
    float mu = sh1[0] / C4;
    float var = sh2[0] / C4 - mu * mu;
    float inv = rsqrtf(var + LN_EPS);
    for (int j = threadIdx.x; j < C4; j += blockDim.x)
        outh[(size_t)row * C4 + j] = (_Float16)((shv[j] - mu) * inv * g[j] + b[j]);
}

// (B,N,d) -> (B,d,H,W)
__global__ void k_feat(const float* __restrict__ tokens, float* __restrict__ out,
                       int N, int d) {
    long long i = (long long)blockIdx.x * blockDim.x + threadIdx.x;
    long long total = (long long)BATCH * N * d;
    if (i >= total) return;
    int c = (int)(i % d); long long t = i / d;
    int n = (int)(t % N); int bb = (int)(t / N);
    out[((size_t)bb * d + c) * N + n] = tokens[((size_t)bb * N + n) * d + c];
}

// ---------------------------------------------------------------------------
// WMMA GEMM: C[M,N] = A[M,K] (f16, row-major) x Wt[N,K]^T (f16) with fused
// bias / BN-affine / GELU / residual epilogue. 128x128 block tile, BK=32,
// 8 waves each computing a 32x64 wave tile via v_wmma_f32_16x16x32_f16.
// ---------------------------------------------------------------------------
#define BM 128
#define BN 128
#define BK 32
#define LDA 40   // BK + 8 halves pad: 80B row stride, conflict-free fan-out

__global__ __launch_bounds__(256)
void k_wmma_gemm(const _Float16* __restrict__ A,
                 const _Float16* __restrict__ Wt,
                 const float* __restrict__ bias,
                 const float* __restrict__ scale,
                 const float* __restrict__ shift,
                 const float* __restrict__ resid,
                 float* __restrict__ C,
                 _Float16* __restrict__ Ch,
                 int M, int N, int K, int do_gelu)
{
    __shared__ _Float16 As[BM * LDA];
    __shared__ _Float16 Bs[BN * LDA];

    const int tid   = threadIdx.x;
    const int lane  = tid & 31;
    const int wid   = tid >> 5;      // 8 waves
    const int wm    = wid & 3;       // 4 along M (32 rows each)
    const int wn    = wid >> 2;      // 2 along N (64 cols each)
    const int lm    = lane & 15;
    const int lhalf = lane >> 4;
    const int klane = lhalf * 8;

    const int bm = blockIdx.y * BM;
    const int bn = blockIdx.x * BN;

    const int ldRow = tid >> 1;          // 0..127
    const int ldK   = (tid & 1) * 16;    // half of a BK row (16 halves = 32B)

    v8f acc[2][4] = {};

    for (int kt = 0; kt < K; kt += BK) {
        // --- stage tiles into LDS (zero-fill out-of-range rows) ---
        {
            uint4 va0 = make_uint4(0, 0, 0, 0), va1 = make_uint4(0, 0, 0, 0);
            int gr = bm + ldRow;
            if (gr < M) {
                const uint4* src = (const uint4*)(A + (size_t)gr * K + kt + ldK);
                va0 = src[0]; va1 = src[1];
                if (kt + BK < K)
                    __builtin_prefetch(A + (size_t)gr * K + kt + BK + ldK, 0, 1);
            }
            uint4* da = (uint4*)(&As[ldRow * LDA + ldK]);
            da[0] = va0; da[1] = va1;

            uint4 vb0 = make_uint4(0, 0, 0, 0), vb1 = make_uint4(0, 0, 0, 0);
            int gn = bn + ldRow;
            if (gn < N) {
                const uint4* src = (const uint4*)(Wt + (size_t)gn * K + kt + ldK);
                vb0 = src[0]; vb1 = src[1];
            }
            uint4* db = (uint4*)(&Bs[ldRow * LDA + ldK]);
            db[0] = vb0; db[1] = vb1;
        }
        __syncthreads();

        // --- build fragments per ISA 16-bit A/B lane layout ---
        v16h af[2], bf[4];
        #pragma unroll
        for (int mi = 0; mi < 2; ++mi) {
            const _Float16* ap = &As[(wm * 32 + mi * 16 + lm) * LDA];
            #pragma unroll
            for (int j = 0; j < 16; ++j) {
                int kk = klane + j + ((j >= 8) ? 8 : 0);
                af[mi][j] = ap[kk];
            }
        }
        #pragma unroll
        for (int ni = 0; ni < 4; ++ni) {
            const _Float16* bp = &Bs[(wn * 64 + ni * 16 + lm) * LDA];
            #pragma unroll
            for (int j = 0; j < 16; ++j) {
                int kk = klane + j + ((j >= 8) ? 8 : 0);
                bf[ni][j] = bp[kk];
            }
        }

        #pragma unroll
        for (int mi = 0; mi < 2; ++mi)
            #pragma unroll
            for (int ni = 0; ni < 4; ++ni)
                acc[mi][ni] = __builtin_amdgcn_wmma_f32_16x16x32_f16(
                    false, af[mi], false, bf[ni], (short)0, acc[mi][ni], false, false);

        __syncthreads();
    }

    // --- fused epilogue; C/D layout: VGPR r -> M = r + 8*(lane>=16), N = lane&15
    #pragma unroll
    for (int mi = 0; mi < 2; ++mi) {
        #pragma unroll
        for (int ni = 0; ni < 4; ++ni) {
            int col = bn + wn * 64 + ni * 16 + lm;
            #pragma unroll
            for (int r = 0; r < 8; ++r) {
                int row = bm + wm * 32 + mi * 16 + 8 * lhalf + r;
                if (row < M && col < N) {
                    float v = acc[mi][ni][r];
                    if (scale)      v = v * scale[col] + shift[col];
                    else if (bias)  v += bias[col];
                    if (do_gelu)    v = gelu_exact(v);
                    if (resid)      v += resid[(size_t)row * N + col];
                    C[(size_t)row * N + col] = v;
                    if (Ch) Ch[(size_t)row * N + col] = (_Float16)v;
                }
            }
        }
    }
}

// ---------------------------------------------------------------------------
// Host orchestration
// ---------------------------------------------------------------------------
static inline char* carve(char*& p, size_t bytes) {
    char* r = p;
    p += (bytes + 255) & ~(size_t)255;
    return r;
}

extern "C" void kernel_launch(void* const* d_in, const int* in_sizes, int n_in,
                              void* d_out, int out_size, void* d_ws, size_t ws_size,
                              hipStream_t stream) {
    (void)in_sizes; (void)n_in; (void)out_size; (void)ws_size;
    int idx = 0;
    const float* x        = (const float*)d_in[idx++];
    const float* embed_w  = (const float*)d_in[idx++];
    const float* embed_b  = (const float*)d_in[idx++];
    const float* ph1_w    = (const float*)d_in[idx++];
    const float* ph1_b    = (const float*)d_in[idx++];
    const float* ph2_w    = (const float*)d_in[idx++];
    const float* ph2_b    = (const float*)d_in[idx++];
    const float* pe_ln_g  = (const float*)d_in[idx++];
    const float* pe_ln_b  = (const float*)d_in[idx++];

    const int DEPTHS[4] = {2, 2, 6, 2};
    struct Blk {
        const float *cc_w, *bn_g, *bn_b, *bn_m, *bn_v, *mix_g, *mix_b,
                    *mlp_g, *mlp_b, *fc1_w, *fc1_b, *fc2_w, *fc2_b;
    };
    Blk blocks[12];
    int nb = 0;
    for (int i = 0; i < 4; ++i)
        for (int j = 0; j < DEPTHS[i]; ++j) {
            Blk bl;
            bl.cc_w  = (const float*)d_in[idx++];
            bl.bn_g  = (const float*)d_in[idx++];
            bl.bn_b  = (const float*)d_in[idx++];
            bl.bn_m  = (const float*)d_in[idx++];
            bl.bn_v  = (const float*)d_in[idx++];
            bl.mix_g = (const float*)d_in[idx++];
            bl.mix_b = (const float*)d_in[idx++];
            bl.mlp_g = (const float*)d_in[idx++];
            bl.mlp_b = (const float*)d_in[idx++];
            bl.fc1_w = (const float*)d_in[idx++];
            bl.fc1_b = (const float*)d_in[idx++];
            bl.fc2_w = (const float*)d_in[idx++];
            bl.fc2_b = (const float*)d_in[idx++];
            blocks[nb++] = bl;
        }
    struct Mrg { const float *ln_g, *ln_b, *red_w; };
    Mrg merges[3];
    for (int i = 0; i < 3; ++i) {
        merges[i].ln_g  = (const float*)d_in[idx++];
        merges[i].ln_b  = (const float*)d_in[idx++];
        merges[i].red_w = (const float*)d_in[idx++];
    }

    const int HW = IMG * IMG;                 // 147456
    const int NP = (IMG / PP) * (IMG / PP);   // 9216
    char* p = (char*)d_ws;
    float*    xe     = (float*)carve(p, (size_t)BATCH * D0 * HW * 4);
    float*    h1     = (float*)carve(p, (size_t)BATCH * D0 * HW * 4);
    _Float16* xen    = (_Float16*)carve(p, (size_t)BATCH * D0 * HW * 2);   // NHWC f16
    float*    pr     = (float*)carve(p, (size_t)BATCH * 3 * NP * 4);
    float*    tokens = (float*)carve(p, (size_t)BATCH * NP * D0 * 4);
    float*    gC     = (float*)carve(p, (size_t)BATCH * NP * D0 * 4 * 4);  // M x 4d f32
    _Float16* A16    = (_Float16*)carve(p, (size_t)BATCH * NP * D0 * 4 * 2);
    _Float16* H16    = (_Float16*)carve(p, (size_t)BATCH * NP * D0 * 4 * 2);
    _Float16* P16    = (_Float16*)carve(p, (size_t)BATCH * NP * D0 * 2 * 2);
    _Float16* W16    = (_Float16*)carve(p, (size_t)3072 * 768 * 2);
    float*    bnsc   = (float*)carve(p, 768 * 4);
    float*    bnsh   = (float*)carve(p, 768 * 4);

    // --- patch embed ---
    {
        int n = BATCH * HW;
        k_embed<<<(n + 255) / 256, 256, 0, stream>>>(x, embed_w, embed_b, xe, xen);
        int nw = D0 * CK;    // 96 x 864
        k_permw<<<(nw + 255) / 256, 256, 0, stream>>>(ph1_w, W16);
        // implicit-GEMM WMMA conv: M = B*HW rows / 128 per block
        k_ph1_wmma<<<(unsigned)(BATCH * HW / CBM), 256, 0, stream>>>(xen, W16, ph1_b, h1);
        int n3 = BATCH * 3 * NP;
        k_ph2<<<(n3 + 255) / 256, 256, 0, stream>>>(h1, ph2_w, ph2_b, pr);
        k_tokens<<<BATCH * NP, 96, 0, stream>>>(xe, pr, pe_ln_g, pe_ln_b, tokens);
    }

    int Hh = IMG / PP, Ww = IMG / PP;
    int Mrows = BATCH * Hh * Ww;
    size_t outOff = 0;
    int bi = 0;
    for (int st = 0; st < 4; ++st) {
        int d = D0 << st;
        for (int j = 0; j < DEPTHS[st]; ++j, ++bi) {
            const Blk& bp = blocks[bi];
            // token-mixing branch: pool -> 1x1 conv (GEMM) + BN + GELU
            long long np = (long long)Mrows * d;
            k_pool<<<(unsigned)((np + 255) / 256), 256, 0, stream>>>(tokens, P16, Hh, Ww, d);
            long long wsz = (long long)d * 2 * d;
            k_f2h<<<(unsigned)((wsz + 255) / 256), 256, 0, stream>>>(bp.cc_w, W16, wsz);
            k_bnprep<<<(d + 255) / 256, 256, 0, stream>>>(bp.bn_g, bp.bn_b, bp.bn_m, bp.bn_v, bnsc, bnsh, d);
            dim3 g1((d + BN - 1) / BN, (Mrows + BM - 1) / BM);
            k_wmma_gemm<<<g1, 256, 0, stream>>>(P16, W16, nullptr, bnsc, bnsh, nullptr,
                                                gC, nullptr, Mrows, d, 2 * d, 1);
            // mix LN + residual (in place on tokens)
            k_ln<<<Mrows, 256, 0, stream>>>(gC, tokens, bp.mix_g, bp.mix_b, tokens, nullptr, d);
            // MLP: LN -> fc1+GELU -> fc2 (+residual)
            k_ln<<<Mrows, 256, 0, stream>>>(tokens, nullptr, bp.mlp_g, bp.mlp_b, nullptr, A16, d);
            wsz = (long long)4 * d * d;
            k_f2h<<<(unsigned)((wsz + 255) / 256), 256, 0, stream>>>(bp.fc1_w, W16, wsz);
            dim3 g2((4 * d + BN - 1) / BN, (Mrows + BM - 1) / BM);
            k_wmma_gemm<<<g2, 256, 0, stream>>>(A16, W16, bp.fc1_b, nullptr, nullptr, nullptr,
                                                gC, H16, Mrows, 4 * d, d, 1);
            k_f2h<<<(unsigned)((wsz + 255) / 256), 256, 0, stream>>>(bp.fc2_w, W16, wsz);
            dim3 g3((d + BN - 1) / BN, (Mrows + BM - 1) / BM);
            k_wmma_gemm<<<g3, 256, 0, stream>>>(H16, W16, bp.fc2_b, nullptr, nullptr, tokens,
                                                tokens, nullptr, Mrows, d, 4 * d, 0);
        }
        // stage feature map output (B,d,H,W)
        long long nf = (long long)Mrows * d;
        k_feat<<<(unsigned)((nf + 255) / 256), 256, 0, stream>>>(tokens, (float*)d_out + outOff,
                                                                 Hh * Ww, d);
        outOff += (size_t)nf;
        if (st < 3) {
            const Mrg& mg = merges[st];
            int M2 = Mrows / 4;
            k_merge_ln<<<M2, 256, 0, stream>>>(tokens, mg.ln_g, mg.ln_b, A16, Hh, Ww, d);
            long long wsz = (long long)2 * d * 4 * d;
            k_f2h<<<(unsigned)((wsz + 255) / 256), 256, 0, stream>>>(mg.red_w, W16, wsz);
            dim3 g4((2 * d + BN - 1) / BN, (M2 + BM - 1) / BM);
            k_wmma_gemm<<<g4, 256, 0, stream>>>(A16, W16, nullptr, nullptr, nullptr, nullptr,
                                                tokens, nullptr, M2, 2 * d, 4 * d, 0);
            Mrows = M2; Hh /= 2; Ww /= 2;
        }
    }
}
